// TopoAMiL_51814485459237
// MI455X (gfx1250) — matrix operand order repaired and checked by reading, not compiled, not run on hardware
//
#include <hip/hip_runtime.h>
#include <hip/hip_bf16.h>
#include <math.h>

// ---------------------------------------------------------------------------
// Problem constants (from reference)
// ---------------------------------------------------------------------------
#define N_INST 8192
#define LDIM   384
#define DDIM   128
#define CCLS   5
#define HDIM   64

typedef __bf16 bf16_t;
typedef __attribute__((ext_vector_type(8)))  __bf16 v8bf;
typedef __attribute__((ext_vector_type(16))) __bf16 v16bf;
typedef __attribute__((ext_vector_type(8)))  float  v8f;

#define WMMA_BF16(A, B, C) \
    __builtin_amdgcn_wmma_f32_16x16x32_bf16(false, (A), false, (B), (short)0, (C), false, false)

// ---------------------------------------------------------------------------
// Kernel 1: per-row  sq[n] = ||x[n]||^2  and split-f32 -> bf16 hi/lo
// ---------------------------------------------------------------------------
__global__ __launch_bounds__(LDIM) void k_split_sq(const float* __restrict__ x,
                                                   bf16_t* __restrict__ hi,
                                                   bf16_t* __restrict__ lo,
                                                   float* __restrict__ sq) {
    __shared__ float red[LDIM];
    const int n = blockIdx.x;
    const int t = threadIdx.x;
    const size_t idx = (size_t)n * LDIM + t;
    const float v = x[idx];
    const bf16_t h = (bf16_t)v;            // RNE to bf16
    const float  hf = (float)h;
    hi[idx] = h;
    lo[idx] = (bf16_t)(v - hf);            // residual fits bf16
    red[t] = v * v;
    __syncthreads();
    if (t < 128) red[t] = red[t] + red[t + 128] + red[t + 256];
    __syncthreads();
    for (int s = 64; s > 0; s >>= 1) {
        if (t < s) red[t] += red[t + s];
        __syncthreads();
    }
    if (t == 0) sq[n] = red[0];
}

// ---------------------------------------------------------------------------
// Kernel 2: attention logits  att[n][c] = tanh(x[n]@aW1 + ab1) @ aW2 + ab2
// ---------------------------------------------------------------------------
__global__ __launch_bounds__(DDIM) void k_att(const float* __restrict__ x,
                                              const float* __restrict__ aW1,
                                              const float* __restrict__ ab1,
                                              const float* __restrict__ aW2,
                                              const float* __restrict__ ab2,
                                              float* __restrict__ att) {
    __shared__ float red[DDIM];
    const int n = blockIdx.x;
    const int d = threadIdx.x;
    const float* __restrict__ xr = x + (size_t)n * LDIM;
    float acc = ab1[d];
#pragma unroll 4
    for (int l = 0; l < LDIM; ++l)
        acc = fmaf(xr[l], aW1[l * DDIM + d], acc);   // aW1 coalesced over d
    const float h = tanhf(acc);
#pragma unroll
    for (int c = 0; c < CCLS; ++c) {
        red[d] = h * aW2[d * CCLS + c];
        __syncthreads();
        for (int s = 64; s > 0; s >>= 1) {
            if (d < s) red[d] += red[d + s];
            __syncthreads();
        }
        if (d == 0) att[(size_t)n * CCLS + c] = red[0] + ab2[c];
        __syncthreads();
    }
}

// ---------------------------------------------------------------------------
// Kernel 3: per-class softmax over N (max, sumexp, normalized weights).
// ---------------------------------------------------------------------------
__global__ __launch_bounds__(1024) void k_softmax(const float* __restrict__ att,
                                                  float* __restrict__ w) {
    __shared__ float red[1024];
    __shared__ float mx[CCLS];
    __shared__ float sm[CCLS];
    const int t = threadIdx.x;

    float lmax[CCLS];
#pragma unroll
    for (int c = 0; c < CCLS; ++c) lmax[c] = -__builtin_inff();
    for (int n = t; n < N_INST; n += 1024) {
#pragma unroll
        for (int c = 0; c < CCLS; ++c)
            lmax[c] = fmaxf(lmax[c], att[(size_t)n * CCLS + c]);
    }
#pragma unroll
    for (int c = 0; c < CCLS; ++c) {
        red[t] = lmax[c];
        __syncthreads();
        for (int s = 512; s > 0; s >>= 1) {
            if (t < s) red[t] = fmaxf(red[t], red[t + s]);
            __syncthreads();
        }
        if (t == 0) mx[c] = red[0];
        __syncthreads();
    }

    float lsum[CCLS];
#pragma unroll
    for (int c = 0; c < CCLS; ++c) lsum[c] = 0.0f;
    for (int n = t; n < N_INST; n += 1024) {
#pragma unroll
        for (int c = 0; c < CCLS; ++c)
            lsum[c] += expf(att[(size_t)n * CCLS + c] - mx[c]);
    }
#pragma unroll
    for (int c = 0; c < CCLS; ++c) {
        red[t] = lsum[c];
        __syncthreads();
        for (int s = 512; s > 0; s >>= 1) {
            if (t < s) red[t] += red[t + s];
            __syncthreads();
        }
        if (t == 0) sm[c] = red[0];
        __syncthreads();
    }

    for (int n = t; n < N_INST; n += 1024) {
#pragma unroll
        for (int c = 0; c < CCLS; ++c)
            w[(size_t)n * CCLS + c] =
                expf(att[(size_t)n * CCLS + c] - mx[c]) / sm[c];
    }
}

// ---------------------------------------------------------------------------
// Kernel 4: bag[c][l] = sum_n w[n][c] * x[n][l]
// ---------------------------------------------------------------------------
__global__ __launch_bounds__(128) void k_bag(const float* __restrict__ x,
                                             const float* __restrict__ w,
                                             float* __restrict__ bag) {
    const int l = blockIdx.x * 128 + threadIdx.x;
    float acc[CCLS] = {0.f, 0.f, 0.f, 0.f, 0.f};
    for (int n = 0; n < N_INST; ++n) {
        const float xv = x[(size_t)n * LDIM + l];
        const float* __restrict__ wr = w + (size_t)n * CCLS;
#pragma unroll
        for (int c = 0; c < CCLS; ++c) acc[c] = fmaf(wr[c], xv, acc[c]);
    }
#pragma unroll
    for (int c = 0; c < CCLS; ++c) bag[c * LDIM + l] = acc[c];
}

// ---------------------------------------------------------------------------
// Kernel 5: per-class heads
// ---------------------------------------------------------------------------
__global__ __launch_bounds__(CCLS * HDIM) void k_heads(const float* __restrict__ bag,
                                                       const float* __restrict__ cW1,
                                                       const float* __restrict__ cb1,
                                                       const float* __restrict__ cW2,
                                                       const float* __restrict__ cb2,
                                                       float* __restrict__ pred) {
    __shared__ float red[CCLS * HDIM];
    const int t = threadIdx.x;
    const int c = t / HDIM;
    const int h = t % HDIM;
    const float* __restrict__ br = bag + c * LDIM;
    const float* __restrict__ w1 = cW1 + (size_t)c * LDIM * HDIM + h;
    float acc = cb1[c * HDIM + h];
#pragma unroll 4
    for (int l = 0; l < LDIM; ++l)
        acc = fmaf(br[l], w1[(size_t)l * HDIM], acc);
    acc = fmaxf(acc, 0.0f) * cW2[c * HDIM + h];
    red[t] = acc;
    __syncthreads();
    for (int s = 32; s > 0; s >>= 1) {
        if ((t & 63) < s) red[t] += red[t + s];
        __syncthreads();
    }
    if (h == 0) pred[c] = red[t] + cb2[c];
}

// ---------------------------------------------------------------------------
// Kernel 6: symmetric Gram GEMM (split-bf16) + fused distance epilogue.
//  - grid 64x64 over 128x128 tiles; blocks with bi<bj exit (lower triangle only)
//  - 512 threads = 16 waves, each owning a 32x32 output tile (4 v8f accums)
//  - A/B hi/lo panels staged in LDS (row stride 40 bf16 -> conflict-free
//    ds_load_b128 fragment reads)
//  - register prefetch of next K-panel overlaps global latency with WMMAs
//    (CU-mode workgroup barrier does not drain LOADcnt -> loads stay in flight)
//  - mirrored tile written via per-wave padded LDS transpose (stride 34),
//    coalesced 64B stores; diagonal blocks skip the mirror
// ---------------------------------------------------------------------------
#define PADK 40                      // padded K-stride of LDS panels (bf16 units)
#define TPAD 34                      // padded row stride of transpose buffer (f32)

__device__ inline v16bf frag_a_lds(const bf16_t* __restrict__ p0, int row0, int lane) {
    const int m  = lane & 15;
    const int ko = (lane >> 4) << 3;        // lanes 0-15: K 0..7/16..23; 16-31: 8..15/24..31
    const bf16_t* p = p0 + (row0 + m) * PADK + ko;
    union { v8bf h[2]; v16bf f; } u;
    u.h[0] = *(const v8bf*)p;
    u.h[1] = *(const v8bf*)(p + 16);
    return u.f;
}

__device__ inline v16bf frag_b_lds(const bf16_t* __restrict__ p0, int col0, int lane) {
    const int n  = lane & 15;
    const int ko = (lane >> 4) << 3;        // lanes 0-15: K 0..7/16..23; 16-31: 8..15/24..31
    const bf16_t* p = p0 + (col0 + n) * PADK + ko;
    union { v8bf h[2]; v16bf f; } u;
    u.h[0] = *(const v8bf*)p;
    u.h[1] = *(const v8bf*)(p + 16);
    return u.f;
}

__global__ __launch_bounds__(512) void k_gram_dist(const bf16_t* __restrict__ hi,
                                                   const bf16_t* __restrict__ lo,
                                                   const float* __restrict__ sq,
                                                   float* __restrict__ dist) {
    const int bi = blockIdx.y;              // row tile
    const int bj = blockIdx.x;              // col tile
    if (bi < bj) return;                    // lower triangle only

    __shared__ __align__(16) char smem[40960];
    bf16_t* sAh = (bf16_t*)smem;            // 128*PADK*2B = 10240 B each
    bf16_t* sAl = sAh + 128 * PADK;
    bf16_t* sBh = sAl + 128 * PADK;
    bf16_t* sBl = sBh + 128 * PADK;

    const int t    = threadIdx.x;
    const int lane = t & 31;
    const int wave = t >> 5;                // 16 waves: 4x4 grid of 32x32 tiles
    const int wr   = (wave >> 2) * 32;
    const int wc   = (wave & 3) * 32;
    const int r0   = bi * 128;
    const int c0   = bj * 128;

    v8f acc[2][2];
    acc[0][0] = (v8f){};  acc[0][1] = (v8f){};
    acc[1][0] = (v8f){};  acc[1][1] = (v8f){};

    // panel copy indices: thread t copies one 16B chunk per panel per step
    const int prow = t >> 2;                // 0..127
    const int pc8  = (t & 3) << 3;          // 0,8,16,24
    const size_t gA = (size_t)(r0 + prow) * LDIM + pc8;
    const size_t gB = (size_t)(c0 + prow) * LDIM + pc8;
    const int    lofs = prow * PADK + pc8;

    // prime the register staging buffers with the k=0 panels
    v8bf rAh = *(const v8bf*)(hi + gA);
    v8bf rAl = *(const v8bf*)(lo + gA);
    v8bf rBh = *(const v8bf*)(hi + gB);
    v8bf rBl = *(const v8bf*)(lo + gB);

    for (int k = 0; k < LDIM; k += 32) {
        // commit staged registers to LDS panels (128 rows x 32 K each)
        *(v8bf*)(sAh + lofs) = rAh;
        *(v8bf*)(sAl + lofs) = rAl;
        *(v8bf*)(sBh + lofs) = rBh;
        *(v8bf*)(sBl + lofs) = rBl;
        __syncthreads();

        // prefetch next K-panel into registers; these loads stay in flight
        // across the WMMA block below (waited at next iteration's ds_store)
        if (k + 32 < LDIM) {
            rAh = *(const v8bf*)(hi + gA + k + 32);
            rAl = *(const v8bf*)(lo + gA + k + 32);
            rBh = *(const v8bf*)(hi + gB + k + 32);
            rBl = *(const v8bf*)(lo + gB + k + 32);
        }

        const v16bf a0h = frag_a_lds(sAh, wr,      lane);
        const v16bf a1h = frag_a_lds(sAh, wr + 16, lane);
        const v16bf a0l = frag_a_lds(sAl, wr,      lane);
        const v16bf a1l = frag_a_lds(sAl, wr + 16, lane);
        const v16bf b0h = frag_b_lds(sBh, wc,      lane);
        const v16bf b1h = frag_b_lds(sBh, wc + 16, lane);
        const v16bf b0l = frag_b_lds(sBl, wc,      lane);
        const v16bf b1l = frag_b_lds(sBl, wc + 16, lane);

        // G ~= hi*hi + hi*lo + lo*hi
        acc[0][0] = WMMA_BF16(a0h, b0h, acc[0][0]);
        acc[0][1] = WMMA_BF16(a0h, b1h, acc[0][1]);
        acc[1][0] = WMMA_BF16(a1h, b0h, acc[1][0]);
        acc[1][1] = WMMA_BF16(a1h, b1h, acc[1][1]);

        acc[0][0] = WMMA_BF16(a0h, b0l, acc[0][0]);
        acc[0][1] = WMMA_BF16(a0h, b1l, acc[0][1]);
        acc[1][0] = WMMA_BF16(a1h, b0l, acc[1][0]);
        acc[1][1] = WMMA_BF16(a1h, b1l, acc[1][1]);

        acc[0][0] = WMMA_BF16(a0l, b0h, acc[0][0]);
        acc[0][1] = WMMA_BF16(a0l, b1h, acc[0][1]);
        acc[1][0] = WMMA_BF16(a1l, b0h, acc[1][0]);
        acc[1][1] = WMMA_BF16(a1l, b1h, acc[1][1]);

        __syncthreads();                    // protect panels for next stage
    }

    // --- direct epilogue: dist(i,j) for this tile -------------------------
    // C/D layout: VGPR v -> M = v + 8*(lane>=16), N = lane&15.
    const int nn = lane & 15;
    const int mb = (lane >> 4) << 3;
#pragma unroll
    for (int ri = 0; ri < 2; ++ri) {
#pragma unroll
        for (int ci = 0; ci < 2; ++ci) {
            const int cidx = c0 + wc + ci * 16 + nn;
            const float sqc = sq[cidx];
            const int rbase = r0 + wr + ri * 16 + mb;
#pragma unroll
            for (int v = 0; v < 8; ++v) {
                const int r = rbase + v;
                float d2 = sq[r] + sqc - 2.0f * acc[ri][ci][v];
                d2 = fmaxf(d2, 0.0f);
                dist[(size_t)r * N_INST + cidx] = sqrtf(d2);
            }
        }
    }

    // --- mirrored epilogue: dist(j,i) via per-wave LDS transpose ----------
    __syncthreads();                        // panels dead; reuse LDS region
    if (bi != bj) {
        float* tb = (float*)smem + wave * (16 * TPAD);   // 16x34 f32 per wave
#pragma unroll
        for (int ri = 0; ri < 2; ++ri) {
            const int gr0 = r0 + wr + ri * 16;           // original rows -> out cols
            const int gc0 = c0 + wc;                     // original cols -> out rows
            // stage raw Gram chunk [m=0..15][n=0..31]
#pragma unroll
            for (int ci = 0; ci < 2; ++ci) {
#pragma unroll
                for (int v = 0; v < 8; ++v)
                    tb[(mb + v) * TPAD + ci * 16 + nn] = acc[ri][ci][v];
            }
            // read transposed: lane -> out col gr0 + (lane&15); two out rows/iter
            const int mloc = lane & 15;
            const int half = lane >> 4;
            const int ocol = gr0 + mloc;
            const float sq_oc = sq[ocol];
#pragma unroll
            for (int j = 0; j < 16; ++j) {
                const int nloc = 2 * j + half;
                const float g = tb[mloc * TPAD + nloc];
                const int orow = gc0 + nloc;
                float d2 = sq[orow] + sq_oc - 2.0f * g;
                d2 = fmaxf(d2, 0.0f);
                dist[(size_t)orow * N_INST + ocol] = sqrtf(d2);
            }
        }
    }
}

// ---------------------------------------------------------------------------
// Launch
// ---------------------------------------------------------------------------
extern "C" void kernel_launch(void* const* d_in, const int* in_sizes, int n_in,
                              void* d_out, int out_size, void* d_ws, size_t ws_size,
                              hipStream_t stream) {
    (void)in_sizes; (void)n_in; (void)out_size; (void)ws_size;

    const float* x   = (const float*)d_in[0];
    const float* aW1 = (const float*)d_in[1];
    const float* ab1 = (const float*)d_in[2];
    const float* aW2 = (const float*)d_in[3];
    const float* ab2 = (const float*)d_in[4];
    const float* cW1 = (const float*)d_in[5];
    const float* cb1 = (const float*)d_in[6];
    const float* cW2 = (const float*)d_in[7];
    const float* cb2 = (const float*)d_in[8];

    // Workspace layout (16B-aligned offsets)
    char* ws = (char*)d_ws;
    const size_t HI_B = (size_t)N_INST * LDIM * sizeof(bf16_t);   // 6,291,456
    bf16_t* hi  = (bf16_t*)ws;
    bf16_t* lo  = (bf16_t*)(ws + HI_B);
    float*  sq  = (float*)(ws + 2 * HI_B);
    float*  att = (float*)(ws + 2 * HI_B + 32768);
    float*  wsm = (float*)(ws + 2 * HI_B + 32768 + 163840);
    float*  bag = (float*)(ws + 2 * HI_B + 32768 + 2 * 163840);

    float* pred = (float*)d_out;           // [1,C] -> 5 floats first
    float* dist = (float*)d_out + CCLS;    // then N*N floats

    // Small path (negligible FLOPs)
    k_split_sq<<<N_INST, LDIM, 0, stream>>>(x, hi, lo, sq);
    k_att<<<N_INST, DDIM, 0, stream>>>(x, aW1, ab1, aW2, ab2, att);
    k_softmax<<<1, 1024, 0, stream>>>(att, wsm);
    k_bag<<<LDIM / 128, 128, 0, stream>>>(x, wsm, bag);
    k_heads<<<1, CCLS * HDIM, 0, stream>>>(bag, cW1, cb1, cW2, cb2, pred);

    // Big path: symmetric Gram + distance (WMMA bf16 split-f32, LDS-staged)
    dim3 grid(N_INST / 128, N_INST / 128);
    k_gram_dist<<<grid, 512, 0, stream>>>(hi, lo, sq, dist);
}
